// SelfAttention_721554506125
// MI455X (gfx1250) — matrix operand order repaired and checked by reading, not compiled, not run on hardware
//
#include <hip/hip_runtime.h>
#include <hip/hip_bf16.h>
#include <math.h>

#define UNITS 512
#define SEQ   2048
#define BATCH 32
#define M_TILE 32

typedef __attribute__((ext_vector_type(16))) __bf16 v16bf;
typedef __attribute__((ext_vector_type(8)))  __bf16 v8bf;
typedef __attribute__((ext_vector_type(8)))  float  v8f;

// fast tanh via v_exp_f32: tanh(x) = (e^2x - 1) / (e^2x + 1), clamped so the
// exponential cannot overflow (|x|>=15 saturates tanh to +-1 in fp32 anyway).
__device__ __forceinline__ float fast_tanhf(float x) {
    float xc = fminf(fmaxf(x, -15.0f), 15.0f);
    float e  = __expf(2.0f * xc);
    return __fdividef(e - 1.0f, e + 1.0f);
}

// ---------------------------------------------------------------------------
// Kernel 0: repack W (512x512 f32, row-major [k][n]) into bf16 laid out in
// this file's WMMA B-fragment order (contiguous K per lane half): fragment (kk, nt) -> 32 lanes x 16 bf16,
// lane L holds K = kk*32 + (L<16?0:16) + h, N = nt*16 + (L&15).
// Main kernel then loads each B fragment as 32 contiguous bytes per lane.
// ---------------------------------------------------------------------------
__global__ __launch_bounds__(256)
void pack_W_bf16(const float* __restrict__ W, __bf16* __restrict__ Wbf) {
    int t    = blockIdx.x * 256 + threadIdx.x;   // 0..16383
    int lane = t & 31;
    int nt   = (t >> 5) & 31;
    int kk   = t >> 10;
    int Kb   = kk * 32 + ((lane & 16) ? 16 : 0);
    int N    = nt * 16 + (lane & 15);
    __bf16* dst = Wbf + (size_t)t * 16;
#pragma unroll
    for (int h = 0; h < 16; ++h)
        dst[h] = (__bf16)W[(size_t)(Kb + h) * UNITS + N];
}

// ---------------------------------------------------------------------------
// Kernel 1: first = s_prev @ U + U_bias   (32 x 512 x 512 -- tiny)
// ---------------------------------------------------------------------------
__global__ __launch_bounds__(512)
void compute_first(const float* __restrict__ s_prev,
                   const float* __restrict__ U,
                   const float* __restrict__ Ub,
                   float* __restrict__ first) {
    int b = blockIdx.x;
    int u = threadIdx.x;                 // 512 threads
    const float* sp = s_prev + b * UNITS;
    float acc = Ub[u];
    for (int k = 0; k < UNITS; ++k)
        acc = fmaf(sp[k], U[(size_t)k * UNITS + u], acc);
    first[b * UNITS + u] = acc;
}

// ---------------------------------------------------------------------------
// Kernel 2: main WMMA kernel.
// Block = 256 threads = 8 waves arranged 2 (M) x 4 (N).
// Block tile: 32 rows of flattened (B*S) x all 512 cols, K = 512.
// score[row] = V . tanh(first[b] + W_bias + hs[row] @ W)
// Scores go to the weights region of d_out (softmax'd in-place by kernel 3).
// ---------------------------------------------------------------------------
__global__ __launch_bounds__(256)
void attn_score_wmma(const float* __restrict__ hs,
                     const float* __restrict__ first,
                     const float* __restrict__ Wb,
                     const float* __restrict__ Vk,
                     const float* __restrict__ Vb,
                     const __bf16* __restrict__ Wbf,
                     float* __restrict__ scores) {
    __shared__ __bf16 lds_a[M_TILE * UNITS];   // 32 KB, A tile as bf16 [m][k]
    __shared__ float  lds_first[UNITS];        // first[b] + W_bias
    __shared__ float  lds_v[UNITS];            // V_kernel
    __shared__ float  lds_score[M_TILE];

    const int tid  = threadIdx.x;
    const int row0 = blockIdx.x * M_TILE;      // flattened (b*S + s) row base
    const int b    = row0 / SEQ;               // SEQ % M_TILE == 0 -> constant b

    // --- stage A tile: 32x512 f32 -> bf16 (coalesced) ---
    const float* hsrc = hs + (size_t)row0 * UNITS;
#pragma unroll 8
    for (int i = 0; i < (M_TILE * UNITS) / 256; ++i) {
        int idx = i * 256 + tid;
        lds_a[idx] = (__bf16)hsrc[idx];
    }
    for (int i = tid; i < UNITS; i += 256) {
        lds_first[i] = first[b * UNITS + i] + Wb[i];
        lds_v[i]     = Vk[i];
    }
    if (tid < M_TILE) lds_score[tid] = 0.0f;
    __syncthreads();

    const int lane    = tid & 31;
    const int wave    = tid >> 5;
    const int wm      = wave >> 2;        // 0..1 : row-half of tile
    const int wn      = wave & 3;         // 0..3 : 128-col slab
    const int l16     = lane & 15;
    const int halfsel = (lane >> 4) & 1;
    const int M       = wm * 16 + l16;    // A row for this lane

    v8f acc[8] = {};                      // 8 n-tiles of 16x16 f32

    // Strength-reduced B pointer: per K-step, this wave's 8 fragments sit at
    // pk + t*512 elements (t*1024 bytes, immediate-offset loads); pk advances
    // by a constant 32*512 elements (32 KB) per K-step.
    const __bf16* pk =
        Wbf + ((size_t)(wn * 8) * 32 + lane) * 16;       // kk = 0 base
    const __bf16* pa_base = lds_a + M * UNITS + halfsel * 16;

    for (int kk = 0; kk < UNITS / 32; ++kk, pk += 32 * UNITS) {
        // A fragment (matches packed B): lanes 0-15 K-off 0, lanes 16-31 K-off 16;
        // VGPRs 0-7 cover K..K+15 contiguously.
        const __bf16* pa = pa_base + kk * 32;
        v8bf alo = *(const v8bf*)pa;          // ds_load_b128
        v8bf ahi = *(const v8bf*)(pa + 8);    // ds_load_b128
        v16bf afrag = __builtin_shufflevector(
            alo, ahi, 0, 1, 2, 3, 4, 5, 6, 7, 8, 9, 10, 11, 12, 13, 14, 15);

        if (kk + 1 < UNITS / 32)   // pull next K-slab of W toward the WGP
            __builtin_prefetch(pk + 32 * UNITS, 0, 3);

        // Two groups of 4 B fragments: group loads issue as one clause and
        // overlap with the previous group's WMMAs; the small live-set keeps
        // the 8 accumulators pinned in place.
#pragma unroll
        for (int g = 0; g < 2; ++g) {
            v16bf bf[4];
#pragma unroll
            for (int t = 0; t < 4; ++t)
                bf[t] = *(const v16bf*)(pk + (size_t)(g * 4 + t) * 512);
#pragma unroll
            for (int t = 0; t < 4; ++t)
                acc[g * 4 + t] = __builtin_amdgcn_wmma_f32_16x16x32_bf16(
                    false, afrag, false, bf[t], (short)0, acc[g * 4 + t],
                    false, false);
        }
    }

    // --- epilogue: tanh + dot with V, reduce over the 512 columns ---
    // C layout: VGPR r -> row = wm*16 + r + halfsel*8 ; lane -> col = l16.
    const int colbase = wn * 128 + l16;
#pragma unroll
    for (int r = 0; r < 8; ++r) {
        float s = 0.0f;
#pragma unroll
        for (int t = 0; t < 8; ++t) {
            int col = colbase + t * 16;
            float x = fast_tanhf(lds_first[col] + acc[t][r]);
            s += x * lds_v[col];
        }
        // reduce across the 16 lanes of this half-wave (same row, 16 cols)
#pragma unroll
        for (int off = 8; off >= 1; off >>= 1)
            s += __shfl_xor(s, off, 16);
        if (l16 == 0) {
            int row = wm * 16 + r + halfsel * 8;
            atomicAdd(&lds_score[row], s);    // ds_add_f32, 4 n-waves combine
        }
    }
    __syncthreads();
    if (tid < M_TILE)
        scores[row0 + tid] = lds_score[tid] + Vb[0];
}

// ---------------------------------------------------------------------------
// Kernel 3: per-batch softmax over S (in-place on d_out weights region) and
// context = (sum_s weights) * s_prev  (faithful to the reference quirk).
// ---------------------------------------------------------------------------
__global__ __launch_bounds__(256)
void softmax_context(const float* __restrict__ s_prev,
                     float* __restrict__ weights,   // holds raw scores on entry
                     float* __restrict__ context) {
    __shared__ float red[256];
    const int b   = blockIdx.x;
    const int tid = threadIdx.x;
    float* wrow = weights + (size_t)b * SEQ;

    float xs[SEQ / 256];
    float m = -1e30f;
#pragma unroll
    for (int i = 0; i < SEQ / 256; ++i) {
        xs[i] = wrow[i * 256 + tid];
        m = fmaxf(m, xs[i]);
    }
    red[tid] = m; __syncthreads();
    for (int st = 128; st > 0; st >>= 1) {
        if (tid < st) red[tid] = fmaxf(red[tid], red[tid + st]);
        __syncthreads();
    }
    m = red[0]; __syncthreads();

    float sum = 0.0f;
#pragma unroll
    for (int i = 0; i < SEQ / 256; ++i) {
        xs[i] = __expf(xs[i] - m);
        sum += xs[i];
    }
    red[tid] = sum; __syncthreads();
    for (int st = 128; st > 0; st >>= 1) {
        if (tid < st) red[tid] += red[tid + st];
        __syncthreads();
    }
    sum = red[0]; __syncthreads();

    const float inv = 1.0f / sum;
    float wsum = 0.0f;
#pragma unroll
    for (int i = 0; i < SEQ / 256; ++i) {
        float w = xs[i] * inv;
        wrow[i * 256 + tid] = w;
        wsum += w;
    }
    red[tid] = wsum; __syncthreads();
    for (int st = 128; st > 0; st >>= 1) {
        if (tid < st) red[tid] += red[tid + st];
        __syncthreads();
    }
    wsum = red[0];

    for (int u = tid; u < UNITS; u += 256)
        context[(size_t)b * UNITS + u] = wsum * s_prev[(size_t)b * UNITS + u];
}

// ---------------------------------------------------------------------------
extern "C" void kernel_launch(void* const* d_in, const int* in_sizes, int n_in,
                              void* d_out, int out_size, void* d_ws, size_t ws_size,
                              hipStream_t stream) {
    const float* s_prev = (const float*)d_in[0];
    const float* hs     = (const float*)d_in[1];
    const float* Wk     = (const float*)d_in[2];
    const float* Wb     = (const float*)d_in[3];
    const float* Uk     = (const float*)d_in[4];
    const float* Ub     = (const float*)d_in[5];
    const float* Vk     = (const float*)d_in[6];
    const float* Vb     = (const float*)d_in[7];

    float* context = (float*)d_out;                    // (B, UNITS)
    float* weights = (float*)d_out + BATCH * UNITS;    // (B, S, 1)

    // workspace: [0, 512KB) bf16-packed W ; [512KB, 576KB) "first" (B x UNITS)
    char*   ws    = (char*)d_ws;
    __bf16* Wbf   = (__bf16*)ws;
    float*  first = (float*)(ws + (size_t)UNITS * UNITS * sizeof(__bf16));

    pack_W_bf16<<<(UNITS * UNITS / 16) / 256, 256, 0, stream>>>(Wk, Wbf);
    compute_first<<<BATCH, UNITS, 0, stream>>>(s_prev, Uk, Ub, first);
    attn_score_wmma<<<(BATCH * SEQ) / M_TILE, 256, 0, stream>>>(
        hs, first, Wb, Vk, Vb, Wbf, weights);
    softmax_context<<<BATCH, 256, 0, stream>>>(s_prev, weights, context);
}
